// ResidualAttentionBlockTensorCache_49065706389553
// MI455X (gfx1250) — compile-verified, matrix operand order checked by reading, not run on hardware
//
#include <hip/hip_runtime.h>

// ---------------------------------------------------------------------------
// Types for CDNA5 WMMA (wave32). 16x16x32 bf16 -> f32 accumulate.
// ---------------------------------------------------------------------------
typedef __attribute__((ext_vector_type(16))) __bf16 v16bf;
typedef __attribute__((ext_vector_type(8)))  float  v8f;

// Native converts: let clang lower f32->bf16 with hardware cvt (RNE).
__device__ __forceinline__ v16bf load_frag_contig(const float* __restrict__ p) {
  // elements 0..7  <- p[0..7]   (K = base .. base+7)
  // elements 8..15 <- p[16..23] (K = base+16 .. base+23)
  const float4* p4 = (const float4*)p;
  float4 a0 = p4[0], a1 = p4[1], b0 = p4[4], b1 = p4[5];
  v16bf r;
  r[0] = (__bf16)a0.x;  r[1] = (__bf16)a0.y;  r[2]  = (__bf16)a0.z;  r[3]  = (__bf16)a0.w;
  r[4] = (__bf16)a1.x;  r[5] = (__bf16)a1.y;  r[6]  = (__bf16)a1.z;  r[7]  = (__bf16)a1.w;
  r[8] = (__bf16)b0.x;  r[9] = (__bf16)b0.y;  r[10] = (__bf16)b0.z;  r[11] = (__bf16)b0.w;
  r[12] = (__bf16)b1.x; r[13] = (__bf16)b1.y; r[14] = (__bf16)b1.z;  r[15] = (__bf16)b1.w;
  return r;
}

__device__ __forceinline__ v16bf load_frag_strided(const float* __restrict__ p, int ld) {
  v16bf r;
#pragma unroll
  for (int e = 0; e < 8; ++e) {
    r[e]     = (__bf16)p[(size_t)e * ld];
    r[e + 8] = (__bf16)p[(size_t)(e + 16) * ld];
  }
  return r;
}

#define WMMA_BF16(A, B, C) \
  __builtin_amdgcn_wmma_f32_16x16x32_bf16(false, (A), false, (B), (short)0, (C), false, false)

__device__ __forceinline__ float gelu_f(float x) {
  const float c0 = 0.7978845608028654f;  // sqrt(2/pi)
  const float c1 = 0.044715f;
  float t = tanhf(c0 * (x + c1 * x * x * x));
  return 0.5f * x * (1.0f + t);
}

// ---------------------------------------------------------------------------
// Problem constants
// ---------------------------------------------------------------------------
#define BB   16
#define QQ   16
#define HH   16
#define DK   64
#define DD   1024
#define LL   1024      // CACHE == CROSS
#define DFF  4096
#define NROWS 256      // B*Q

// ---------------------------------------------------------------------------
// float4 grid-stride copy (KV-cache passthrough into output tuple)
// ---------------------------------------------------------------------------
__global__ __launch_bounds__(256)
void copy_f4_kernel(const float4* __restrict__ src, float4* __restrict__ dst, long n4) {
  long i = (long)blockIdx.x * blockDim.x + threadIdx.x;
  long stride = (long)gridDim.x * blockDim.x;
  for (; i < n4; i += stride) dst[i] = src[i];
}

// Scatter freshly projected K/V rows into the last Q slots of each batch.
__global__ __launch_bounds__(256)
void scatter_kv_kernel(const float* __restrict__ kn, const float* __restrict__ vn,
                       float* __restrict__ kc, float* __restrict__ vc) {
  int idx = blockIdx.x * 256 + threadIdx.x;       // over 256*1024
  if (idx >= NROWS * DD) return;
  int row = idx >> 10, col = idx & 1023;
  int b = row >> 4, qi = row & 15;
  size_t d = ((size_t)b * LL + (LL - QQ) + qi) * DD + col;
  kc[d] = kn[idx];
  vc[d] = vn[idx];
}

// ---------------------------------------------------------------------------
// LayerNorm: one block per row of 1024
// ---------------------------------------------------------------------------
__global__ __launch_bounds__(256)
void ln_kernel(const float* __restrict__ x, const float* __restrict__ g,
               const float* __restrict__ be, float* __restrict__ o) {
  __shared__ float red[256];
  int row = blockIdx.x;
  const float* xr = x + (size_t)row * DD;
  float s = 0.f;
  for (int i = threadIdx.x; i < DD; i += 256) s += xr[i];
  red[threadIdx.x] = s; __syncthreads();
  for (int off = 128; off; off >>= 1) {
    if ((int)threadIdx.x < off) red[threadIdx.x] += red[threadIdx.x + off];
    __syncthreads();
  }
  float mean = red[0] * (1.0f / DD); __syncthreads();
  float s2 = 0.f;
  for (int i = threadIdx.x; i < DD; i += 256) { float d = xr[i] - mean; s2 += d * d; }
  red[threadIdx.x] = s2; __syncthreads();
  for (int off = 128; off; off >>= 1) {
    if ((int)threadIdx.x < off) red[threadIdx.x] += red[threadIdx.x + off];
    __syncthreads();
  }
  float inv = rsqrtf(red[0] * (1.0f / DD) + 1e-5f);
  float* orow = o + (size_t)row * DD;
  for (int i = threadIdx.x; i < DD; i += 256)
    orow[i] = (xr[i] - mean) * inv * g[i] + be[i];
}

// ---------------------------------------------------------------------------
// Linear GEMM with WMMA:  C[256,N] = A[256,K] * W[K,N]  (+epilogue)
// One 16x64 strip per wave: A fragment loaded once per K-step, reused by
// 4 WMMAs (4 accumulators). 4 waves / block.
// ---------------------------------------------------------------------------
enum { EPI_BIAS = 0, EPI_BIAS_GELU = 1, EPI_BIAS_RESID = 2 };

template <int EPI>
__global__ __launch_bounds__(128)
void gemm_linear_kernel(const float* __restrict__ A, const float* __restrict__ W,
                        const float* __restrict__ bias, const float* __restrict__ resid,
                        float* __restrict__ C, int N, int K) {
  const int lane = threadIdx.x & 31;
  const int lrow = lane & 15;
  const int kb   = (lane >> 4) << 3;         // 0 or 8
  int t = blockIdx.x * 4 + (threadIdx.x >> 5);
  const int nt64 = N >> 6;                   // 64-wide strips along N
  int mt = t / nt64, nb = t - mt * nt64;
  if (mt >= NROWS / 16) return;              // wave-uniform guard

  const float* Abase = A + (size_t)(mt * 16 + lrow) * K;
  const int ncol0 = nb * 64 + lrow;

  v8f acc0 = {}, acc1 = {}, acc2 = {}, acc3 = {};
  for (int k0 = 0; k0 < K; k0 += 32) {
    __builtin_prefetch(Abase + k0 + kb + 32, 0, 1);
    v16bf af = load_frag_contig(Abase + k0 + kb);
    const float* pw = W + (size_t)(k0 + kb) * N + ncol0;
    v16bf b0 = load_frag_strided(pw,      N);
    v16bf b1 = load_frag_strided(pw + 16, N);
    v16bf b2 = load_frag_strided(pw + 32, N);
    v16bf b3 = load_frag_strided(pw + 48, N);
    acc0 = WMMA_BF16(af, b0, acc0);
    acc1 = WMMA_BF16(af, b1, acc1);
    acc2 = WMMA_BF16(af, b2, acc2);
    acc3 = WMMA_BF16(af, b3, acc3);
  }

  const int cn = lane & 15;
  const int mb = (lane >> 4) << 3;
  v8f accs[4] = {acc0, acc1, acc2, acc3};
#pragma unroll
  for (int j = 0; j < 4; ++j) {
#pragma unroll
    for (int v = 0; v < 8; ++v) {
      int m = mt * 16 + mb + v;
      int n = nb * 64 + j * 16 + cn;
      float r = accs[j][v] + bias[n];
      if (EPI == EPI_BIAS_GELU)  r = gelu_f(r);
      if (EPI == EPI_BIAS_RESID) r += resid[(size_t)m * N + n];
      C[(size_t)m * N + n] = r;
    }
  }
}

// ---------------------------------------------------------------------------
// Attention scores: per (b,h): S[16,1024] = Q[16,64] * K^T, scaled by 1/8.
// One 16x64 key strip per wave (4 accumulators, Q fragment reused).
// Tiles = B*H*(L/64) = 4096.
// ---------------------------------------------------------------------------
__global__ __launch_bounds__(128)
void attn_scores_kernel(const float* __restrict__ qm, const float* __restrict__ km,
                        float* __restrict__ sc) {
  const int lane = threadIdx.x & 31;
  const int lrow = lane & 15;
  const int kb   = (lane >> 4) << 3;
  int t = blockIdx.x * 4 + (threadIdx.x >> 5);   // 0..4095
  int b  = t >> 8;
  int h  = (t >> 4) & 15;
  int nt = t & 15;                               // 64-wide key strip

  const float* Abase = qm + ((size_t)b * QQ + lrow) * DD + h * DK;
  const float* Bb0 = km + ((size_t)b * LL + nt * 64 +  0 + lrow) * DD + h * DK;
  const float* Bb1 = km + ((size_t)b * LL + nt * 64 + 16 + lrow) * DD + h * DK;
  const float* Bb2 = km + ((size_t)b * LL + nt * 64 + 32 + lrow) * DD + h * DK;
  const float* Bb3 = km + ((size_t)b * LL + nt * 64 + 48 + lrow) * DD + h * DK;

  v8f acc0 = {}, acc1 = {}, acc2 = {}, acc3 = {};
#pragma unroll
  for (int k0 = 0; k0 < DK; k0 += 32) {
    v16bf af = load_frag_contig(Abase + k0 + kb);
    acc0 = WMMA_BF16(af, load_frag_contig(Bb0 + k0 + kb), acc0);
    acc1 = WMMA_BF16(af, load_frag_contig(Bb1 + k0 + kb), acc1);
    acc2 = WMMA_BF16(af, load_frag_contig(Bb2 + k0 + kb), acc2);
    acc3 = WMMA_BF16(af, load_frag_contig(Bb3 + k0 + kb), acc3);
  }

  const float scale = 0.125f;   // 1/sqrt(64)
  const int cn = lane & 15;
  const int mb = (lane >> 4) << 3;
  v8f accs[4] = {acc0, acc1, acc2, acc3};
#pragma unroll
  for (int j = 0; j < 4; ++j) {
    float* crow = sc + ((size_t)(b * HH + h) * QQ) * LL + nt * 64 + j * 16;
#pragma unroll
    for (int v = 0; v < 8; ++v)
      crow[(size_t)(mb + v) * LL + cn] = accs[j][v] * scale;
  }
}

// ---------------------------------------------------------------------------
// Masked softmax over L=1024, one block per (b,h,q) row; in-place on scores.
// ---------------------------------------------------------------------------
__global__ __launch_bounds__(256)
void softmax_kernel(float* __restrict__ sc, const unsigned char* __restrict__ mask) {
  __shared__ float red[256];
  int row = blockIdx.x;                // (b*H + h)*Q + q
  int q = row & 15;
  int b = row >> 8;                    // H*Q = 256 rows per batch
  float* s = sc + (size_t)row * LL;
  const unsigned char* mk = mask + ((size_t)(b * QQ + q)) * LL;
  const float NEG = -3.402823466e38f;

  float mx = NEG;
  for (int i = threadIdx.x; i < LL; i += 256) {
    float v = mk[i] ? s[i] : NEG;
    mx = fmaxf(mx, v);
  }
  red[threadIdx.x] = mx; __syncthreads();
  for (int off = 128; off; off >>= 1) {
    if ((int)threadIdx.x < off)
      red[threadIdx.x] = fmaxf(red[threadIdx.x], red[threadIdx.x + off]);
    __syncthreads();
  }
  mx = red[0]; __syncthreads();

  float sum = 0.f;
  for (int i = threadIdx.x; i < LL; i += 256) {
    float v = mk[i] ? __expf(s[i] - mx) : 0.f;
    s[i] = v;
    sum += v;
  }
  red[threadIdx.x] = sum; __syncthreads();
  for (int off = 128; off; off >>= 1) {
    if ((int)threadIdx.x < off) red[threadIdx.x] += red[threadIdx.x + off];
    __syncthreads();
  }
  float inv = 1.0f / red[0];
  for (int i = threadIdx.x; i < LL; i += 256) s[i] *= inv;
}

// ---------------------------------------------------------------------------
// Attention output: per (b,h): O[16,64] = P[16,1024] * V[1024, head slice].
// One whole head per wave (4 accumulators cover DK=64). Tiles = B*H = 256.
// ---------------------------------------------------------------------------
__global__ __launch_bounds__(128)
void attn_out_kernel(const float* __restrict__ pm, const float* __restrict__ vm,
                     float* __restrict__ om) {
  const int lane = threadIdx.x & 31;
  const int lrow = lane & 15;
  const int kb   = (lane >> 4) << 3;
  int t = blockIdx.x * 4 + (threadIdx.x >> 5);   // 0..255
  int b  = t >> 4;
  int h  = t & 15;

  const float* Abase = pm + ((size_t)(b * HH + h) * QQ + lrow) * LL;
  const float* Vb = vm + (size_t)b * LL * DD + h * DK + lrow;

  v8f acc0 = {}, acc1 = {}, acc2 = {}, acc3 = {};
  for (int k0 = 0; k0 < LL; k0 += 32) {
    __builtin_prefetch(Abase + k0 + kb + 32, 0, 1);
    v16bf af = load_frag_contig(Abase + k0 + kb);
    const float* pv = Vb + (size_t)(k0 + kb) * DD;
    acc0 = WMMA_BF16(af, load_frag_strided(pv,      DD), acc0);
    acc1 = WMMA_BF16(af, load_frag_strided(pv + 16, DD), acc1);
    acc2 = WMMA_BF16(af, load_frag_strided(pv + 32, DD), acc2);
    acc3 = WMMA_BF16(af, load_frag_strided(pv + 48, DD), acc3);
  }

  const int cn = lane & 15;
  const int mb = (lane >> 4) << 3;
  v8f accs[4] = {acc0, acc1, acc2, acc3};
#pragma unroll
  for (int j = 0; j < 4; ++j) {
    float* crow = om + (size_t)(b * QQ) * DD + h * DK + j * 16;
#pragma unroll
    for (int v = 0; v < 8; ++v)
      crow[(size_t)(mb + v) * DD + cn] = accs[j][v];
  }
}

// ---------------------------------------------------------------------------
// Launch
// ---------------------------------------------------------------------------
extern "C" void kernel_launch(void* const* d_in, const int* in_sizes, int n_in,
                              void* d_out, int out_size, void* d_ws, size_t ws_size,
                              hipStream_t stream) {
  (void)in_sizes; (void)n_in; (void)out_size; (void)ws_size;

  const float* x        = (const float*)d_in[0];
  const float* k_in     = (const float*)d_in[1];
  const float* v_in     = (const float*)d_in[2];
  const float* cross_k  = (const float*)d_in[3];
  const float* cross_v  = (const float*)d_in[4];
  const unsigned char* self_mask  = (const unsigned char*)d_in[5];
  const unsigned char* cross_mask = (const unsigned char*)d_in[6];
  const float* ln1_g = (const float*)d_in[7];
  const float* ln1_b = (const float*)d_in[8];
  const float* ln2_g = (const float*)d_in[9];
  const float* ln2_b = (const float*)d_in[10];
  const float* ln3_g = (const float*)d_in[11];
  const float* ln3_b = (const float*)d_in[12];
  const float* wq_s = (const float*)d_in[13];
  const float* bq_s = (const float*)d_in[14];
  const float* wk_s = (const float*)d_in[15];
  const float* bk_s = (const float*)d_in[16];
  const float* wv_s = (const float*)d_in[17];
  const float* bv_s = (const float*)d_in[18];
  const float* wo_s = (const float*)d_in[19];
  const float* bo_s = (const float*)d_in[20];
  const float* wq_c = (const float*)d_in[21];
  const float* bq_c = (const float*)d_in[22];
  const float* wo_c = (const float*)d_in[23];
  const float* bo_c = (const float*)d_in[24];
  const float* w1   = (const float*)d_in[25];
  const float* b1   = (const float*)d_in[26];
  const float* w2   = (const float*)d_in[27];
  const float* b2   = (const float*)d_in[28];

  // Output tuple: x[256*1024], k_cache[16M], v_cache[16M]
  float* out_x = (float*)d_out;
  float* kco   = out_x + (size_t)NROWS * DD;
  float* vco   = kco + (size_t)BB * LL * DD;

  // Workspace layout (floats), ~23 MB total
  float* ws   = (float*)d_ws;
  float* xn   = ws;                        // 256*1024
  float* qbuf = xn   + (size_t)NROWS * DD;
  float* knew = qbuf + (size_t)NROWS * DD;
  float* vnew = knew + (size_t)NROWS * DD;
  float* sc   = vnew + (size_t)NROWS * DD; // 16*16*16*1024 = 4M floats
  float* abuf = sc   + (size_t)BB * HH * QQ * LL;
  float* xbuf = abuf + (size_t)NROWS * DD;
  float* hbuf = sc;                        // reuse scores buffer for MLP hidden

  const long cache_n4 = (long)BB * LL * DD / 4;

  // ---- KV-cache passthrough ----
  copy_f4_kernel<<<2048, 256, 0, stream>>>((const float4*)k_in, (float4*)kco, cache_n4);
  copy_f4_kernel<<<2048, 256, 0, stream>>>((const float4*)v_in, (float4*)vco, cache_n4);

  // ---- self-attention ----
  ln_kernel<<<NROWS, 256, 0, stream>>>(x, ln1_g, ln1_b, xn);
  gemm_linear_kernel<EPI_BIAS><<<64, 128, 0, stream>>>(xn, wq_s, bq_s, nullptr, qbuf, DD, DD);
  gemm_linear_kernel<EPI_BIAS><<<64, 128, 0, stream>>>(xn, wk_s, bk_s, nullptr, knew, DD, DD);
  gemm_linear_kernel<EPI_BIAS><<<64, 128, 0, stream>>>(xn, wv_s, bv_s, nullptr, vnew, DD, DD);
  scatter_kv_kernel<<<1024, 256, 0, stream>>>(knew, vnew, kco, vco);
  attn_scores_kernel<<<1024, 128, 0, stream>>>(qbuf, kco, sc);
  softmax_kernel<<<4096, 256, 0, stream>>>(sc, self_mask);
  attn_out_kernel<<<64, 128, 0, stream>>>(sc, vco, abuf);
  gemm_linear_kernel<EPI_BIAS_RESID><<<64, 128, 0, stream>>>(abuf, wo_s, bo_s, x, xbuf, DD, DD);

  // ---- cross-attention ----
  ln_kernel<<<NROWS, 256, 0, stream>>>(xbuf, ln2_g, ln2_b, xn);
  gemm_linear_kernel<EPI_BIAS><<<64, 128, 0, stream>>>(xn, wq_c, bq_c, nullptr, qbuf, DD, DD);
  attn_scores_kernel<<<1024, 128, 0, stream>>>(qbuf, cross_k, sc);
  softmax_kernel<<<4096, 256, 0, stream>>>(sc, cross_mask);
  attn_out_kernel<<<64, 128, 0, stream>>>(sc, cross_v, abuf);
  gemm_linear_kernel<EPI_BIAS_RESID><<<64, 128, 0, stream>>>(abuf, wo_c, bo_c, xbuf, xbuf, DD, DD);

  // ---- MLP ----
  ln_kernel<<<NROWS, 256, 0, stream>>>(xbuf, ln3_g, ln3_b, xn);
  gemm_linear_kernel<EPI_BIAS_GELU><<<256, 128, 0, stream>>>(xn, w1, b1, nullptr, hbuf, DFF, DD);
  gemm_linear_kernel<EPI_BIAS_RESID><<<64, 128, 0, stream>>>(hbuf, w2, b2, xbuf, out_x, DD, DFF);
}
